// InteractiveAttention_6296422056375
// MI455X (gfx1250) — matrix operand order repaired and checked by reading, not compiled, not run on hardware
//
#include <hip/hip_runtime.h>
#include <math.h>

// ---------------------------------------------------------------------------
// InteractiveAttention for MI455X (gfx1250, wave32, WMMA).
// bf16 end-to-end dataflow; GEMM K-loop is double-buffered with
// GLOBAL_LOAD_ASYNC_TO_LDS_B128 (ASYNCcnt) so copies overlap WMMA.
// ---------------------------------------------------------------------------

typedef __bf16 bf16;
typedef bf16         bf16x16 __attribute__((ext_vector_type(16)));
typedef float        floatx8 __attribute__((ext_vector_type(8)));
typedef unsigned int uintx8  __attribute__((ext_vector_type(8)));
typedef int          v4i     __attribute__((vector_size(16)));   // builtin's pointee

constexpr int BSZ   = 4;
constexpr int LEN   = 1024;
constexpr int DIM   = 1024;
constexpr int NH    = 16;
constexpr int HDIM  = 64;          // DIM / NH
constexpr int MROWS = BSZ * LEN;   // 4096

__device__ __forceinline__ unsigned bfbits(float f) {
  union { float f; unsigned u; } v; v.f = f;
  return (v.u + 0x7FFFu + ((v.u >> 16) & 1u)) >> 16;   // RNE
}
__device__ __forceinline__ unsigned pk2bf(float lo, float hi) {
  return (bfbits(hi) << 16) | (bfbits(lo) & 0xFFFFu);
}
__device__ __forceinline__ bf16 f2bf(float f) {
  unsigned short s = (unsigned short)bfbits(f);
  return __builtin_bit_cast(bf16, s);
}
__device__ __forceinline__ float bf2f_lo(unsigned u) {
  union { unsigned x; float f; } v; v.x = u << 16; return v.f;
}
__device__ __forceinline__ float bf2f_hi(unsigned u) {
  union { unsigned x; float f; } v; v.x = u & 0xFFFF0000u; return v.f;
}
__device__ __forceinline__ float fast_sigmoid(float x) {
  return __builtin_amdgcn_rcpf(1.0f + __expf(-x));
}

// --------------------------- async LDS staging -----------------------------
#if __has_builtin(__builtin_amdgcn_global_load_async_to_lds_b128)
#define HAVE_ASYNC_LDS 1
__device__ __forceinline__ void async_b128(const void* g, void* l) {
  __builtin_amdgcn_global_load_async_to_lds_b128(
      (__attribute__((address_space(1))) v4i*)(g),
      (__attribute__((address_space(3))) v4i*)(l), 0, 0);
}
template <int N>
__device__ __forceinline__ void async_wait_le() {
#if __has_builtin(__builtin_amdgcn_s_wait_asynccnt)
  __builtin_amdgcn_s_wait_asynccnt(N);
#else
  asm volatile("s_wait_asynccnt %0" :: "i"(N) : "memory");
#endif
}
#else
#define HAVE_ASYNC_LDS 0
__device__ __forceinline__ void async_b128(const void* g, void* l) {
  *(uint4*)l = *(const uint4*)g;    // fallback: b128 load + b128 ds store
}
template <int N> __device__ __forceinline__ void async_wait_le() {}
#endif

// fp32 -> bf16 pack, 8 elements per thread (n % 2048 == 0).
__global__ __launch_bounds__(256)
void pack_bf16_kernel(const float* __restrict__ x, bf16* __restrict__ y)
{
  int i = blockIdx.x * 256 + threadIdx.x;
  const float4* x4 = (const float4*)x;
  float4 a = x4[2 * i], b = x4[2 * i + 1];
  ((uint4*)y)[i] = make_uint4(pk2bf(a.x, a.y), pk2bf(a.z, a.w),
                              pk2bf(b.x, b.y), pk2bf(b.z, b.w));
}

// ---------------------------------------------------------------------------
// WMMA GEMM:  C[M,N] = ACT( concatK(A0,A1)[M,K] @ W[K,N] + bias[N] )
//   A0/A1/W are bf16; plain GEMM: A1==A0, K0==K.
//   ACT: 0 none, 1 sigmoid, 2 relu.  BF16OUT: C is bf16 (else fp32).
// Block tile 128x128, K-tile 32, 8 waves; wave tile 32x64 (2x4 accumulators).
// Double-buffered LDS: async copies for tile i+1 fly under tile i's WMMAs.
// ---------------------------------------------------------------------------
template <int ACT, int BF16OUT>
__global__ __launch_bounds__(256)
void gemm_bf16_wmma(const bf16* __restrict__ A0, const bf16* __restrict__ A1,
                    int K0,
                    const bf16* __restrict__ W, const float* __restrict__ bias,
                    void* __restrict__ Cout, int N, int K)
{
  const int tid  = threadIdx.x;
  const int wid  = tid >> 5;
  const int lane = tid & 31;
  const int rowBase = blockIdx.y * 128;
  const int colBase = blockIdx.x * 128;
  const int waveM = (wid >> 1) * 32;
  const int waveN = (wid & 1) * 64;
  const int K1 = K - K0;

  // Packed bf16-pair tiles; 16B-aligned row strides for b128 ops.
  __shared__ unsigned sA[2][128][20];  // 2 x 128 x (16+4) uints, 80B rows
  __shared__ unsigned sB[2][32][68];   // 2 x 32 x (64+4) uints, 272B rows

  floatx8 acc[2][4];
#pragma unroll
  for (int mi = 0; mi < 2; ++mi)
#pragma unroll
    for (int ni = 0; ni < 4; ++ni)
#pragma unroll
      for (int e = 0; e < 8; ++e) acc[mi][ni][e] = 0.0f;

  // Staging coordinates: A: 2 thr/row x 16 bf16; B: 8 thr/row x 16 bf16.
  const int ar = tid >> 1, aq = tid & 1;
  const int br = tid >> 3, bq = tid & 7;

  // Issue 4 async b128 copies (per thread) for K-tile `ktv` into buffer `buf`.
#define STAGE_TILE(buf, ktv)                                                  \
  {                                                                           \
    int gk_ = (ktv) + 16 * aq;                      /* K0 % 16 == 0 */        \
    const bf16* asrc_ = (gk_ < K0)                                            \
        ? A0 + (size_t)(rowBase + ar) * K0 + gk_                              \
        : A1 + (size_t)(rowBase + ar) * K1 + (gk_ - K0);                      \
    const bf16* bsrc_ = W + (size_t)((ktv) + br) * N + colBase + 16 * bq;     \
    async_b128(asrc_,     &sA[buf][ar][8 * aq]);                              \
    async_b128(asrc_ + 8, &sA[buf][ar][8 * aq + 4]);                          \
    async_b128(bsrc_,     &sB[buf][br][8 * bq]);                              \
    async_b128(bsrc_ + 8, &sB[buf][br][8 * bq + 4]);                          \
  }

  STAGE_TILE(0, 0);                                  // prologue

  for (int kt = 0; kt < K; kt += 32) {
    const int cur = (kt >> 5) & 1;
    if (kt + 32 < K) {                               // pipeline next tile
      STAGE_TILE(cur ^ 1, kt + 32);
      async_wait_le<4>();                            // tile `cur` landed
    } else {
      async_wait_le<0>();
    }
    __syncthreads();                                 // all waves' copies visible

    // Fragments (ISA 7.12.2 wave32 layouts); A = 2x ds_load_b128, B = 16 strided b16 loads.
    const int half4 = (lane >> 4) * 4;
    bf16x16 af[2], bfr[4];
#pragma unroll
    for (int mi = 0; mi < 2; ++mi) {
      const unsigned* arow = &sA[cur][waveM + 16 * mi + (lane & 15)][0];
      uintx8 u;
#pragma unroll
      for (int v = 0; v < 4; ++v) { u[v] = arow[half4 + v]; u[4 + v] = arow[8 + half4 + v]; }
      af[mi] = __builtin_bit_cast(bf16x16, u);
    }
#pragma unroll
    for (int ni = 0; ni < 4; ++ni) {                 // B: lane = N col, elems = K (table order)
      const bf16* bcol = (const bf16*)&sB[cur][0][0] + waveN + 16 * ni + (lane & 15);
      bf16x16 t;
#pragma unroll
      for (int s = 0; s < 16; ++s) t[s] = bcol[(s < 8 ? 2 * half4 + s : 16 + 2 * half4 + s - 8) * 136];
      bfr[ni] = t;
    }
#pragma unroll
    for (int mi = 0; mi < 2; ++mi)
#pragma unroll
      for (int ni = 0; ni < 4; ++ni)
        acc[mi][ni] = __builtin_amdgcn_wmma_f32_16x16x32_bf16(
            false, af[mi], false, bfr[ni], (short)0, acc[mi][ni], false, false);

    __syncthreads();     // buffer `cur` free for overwrite next iteration
  }
#undef STAGE_TILE

  // Epilogue: lane 0..15: M=r,N=lane; lane 16..31: M=8+r,N=lane-16.
  const int col = lane & 15, rowOff = (lane >> 4) * 8;
#pragma unroll
  for (int mi = 0; mi < 2; ++mi)
#pragma unroll
    for (int ni = 0; ni < 4; ++ni) {
      int gcol = colBase + waveN + 16 * ni + col;
      float bv = bias[gcol];
#pragma unroll
      for (int r = 0; r < 8; ++r) {
        int grow = rowBase + waveM + 16 * mi + rowOff + r;
        float v = acc[mi][ni][r] + bv;
        if (ACT == 1) v = fast_sigmoid(v);
        if (ACT == 2) v = fmaxf(v, 0.0f);
        if (BF16OUT) ((bf16*)Cout)[(size_t)grow * N + gcol] = f2bf(v);
        else         ((float*)Cout)[(size_t)grow * N + gcol] = v;
      }
    }
}

// ---------------------------------------------------------------------------
// Flash attention over bf16 Q/K/V; O (bf16) may alias Q.
// One block = one (b,h) + 128 query rows; 8 waves x 16 rows.
// 1/sqrt(hd) folded into softmax (scores kept unscaled in LDS).
// ---------------------------------------------------------------------------
__global__ __launch_bounds__(256)
void flash_attn_wmma(const bf16* __restrict__ Q, const bf16* __restrict__ Km,
                     const bf16* __restrict__ V, bf16* __restrict__ O)
{
  const int tid  = threadIdx.x;
  const int wid  = tid >> 5;
  const int lane = tid & 31;
  const int b    = blockIdx.y / NH;
  const int h    = blockIdx.y % NH;
  const int qBase = blockIdx.x * 128;
  const size_t headBase = (size_t)b * LEN * DIM + (size_t)h * HDIM;

  __shared__ unsigned sQ[128][36];      // 128 x (32+4) uints, 144B rows
  __shared__ bf16     sKt[HDIM][40];    // transposed K tile [hd][key], 80B rows
  __shared__ unsigned sV[32][36];       // 32 x (32+4) uints, 144B rows
  __shared__ float    sS[8][16][36];    // per-wave score scratch
  __shared__ unsigned sP[8][16][20];    // per-wave packed P, 80B rows
  __shared__ float    rowM[8][16], rowL[8][16], rowScale[8][16];

  { // stage Q: 4x b128 chunks per thread (pure copy, bf16 already)
    int r = tid >> 1, qh = tid & 1;
    const bf16* qsrc = Q + headBase + (size_t)(qBase + r) * DIM + 32 * qh;
#pragma unroll
    for (int j = 0; j < 4; ++j)
      async_b128(qsrc + 8 * j, &sQ[r][16 * qh + 4 * j]);
    async_wait_le<0>();
  }
  if (lane < 16) { rowM[wid][lane] = -3.0e38f; rowL[wid][lane] = 0.0f; }
  __syncthreads();

  // Q fragments: two 16x32 A-fragments (k-steps over hd).
  const int half4 = (lane >> 4) * 4;
  bf16x16 qf[2];
  {
    const unsigned* qrow = &sQ[wid * 16 + (lane & 15)][0];
#pragma unroll
    for (int ks = 0; ks < 2; ++ks) {
      uintx8 u;
#pragma unroll
      for (int v = 0; v < 4; ++v) {
        u[v]     = qrow[16 * ks + half4 + v];
        u[4 + v] = qrow[16 * ks + 8 + half4 + v];
      }
      qf[ks] = __builtin_bit_cast(bf16x16, u);
    }
  }

  floatx8 Oacc[4];
#pragma unroll
  for (int t = 0; t < 4; ++t)
#pragma unroll
    for (int e = 0; e < 8; ++e) Oacc[t][e] = 0.0f;

  const int col = lane & 15, rowOff = (lane >> 4) * 8;

  for (int kt = 0; kt < LEN; kt += 32) {
    __syncthreads();
    { // V: async b128 copy; K: transpose via b16 scatter (no conversion)
      int r = tid >> 3, q8 = tid & 7;
      async_b128(V + headBase + (size_t)(kt + r) * DIM + 8 * q8, &sV[r][4 * q8]);
      union { uint4 v; unsigned short us[8]; } kk;
      kk.v = *(const uint4*)(Km + headBase + (size_t)(kt + r) * DIM + 8 * q8);
      int c0 = 8 * q8;
#pragma unroll
      for (int i = 0; i < 8; ++i)
        sKt[c0 + i][r] = __builtin_bit_cast(bf16, kk.us[i]);
      async_wait_le<0>();
    }
    __syncthreads();

    // S(16x32) = Q(16x64) @ K^T : 2 N-tiles x 2 K-steps.
    floatx8 sc[2];
#pragma unroll
    for (int ni = 0; ni < 2; ++ni) {
#pragma unroll
      for (int e = 0; e < 8; ++e) sc[ni][e] = 0.0f;
#pragma unroll
      for (int ks = 0; ks < 2; ++ks) {
        const bf16* kcol = &sKt[32 * ks][16 * ni + (lane & 15)];
        bf16x16 bfr;
#pragma unroll
        for (int s = 0; s < 16; ++s) bfr[s] = kcol[(s < 8 ? 2 * half4 + s : 16 + 2 * half4 + s - 8) * 40];
        sc[ni] = __builtin_amdgcn_wmma_f32_16x16x32_bf16(
            false, qf[ks], false, bfr, (short)0, sc[ni], false, false);
      }
    }
#pragma unroll
    for (int ni = 0; ni < 2; ++ni)
#pragma unroll
      for (int r = 0; r < 8; ++r)
        sS[wid][rowOff + r][16 * ni + col] = sc[ni][r];
    __syncthreads();

    // Online softmax (scale 0.125 applied here); lane r < 16 owns row r.
    if (lane < 16) {
      float mprev = rowM[wid][lane];
      float mmax = mprev;
      for (int c = 0; c < 32; ++c)
        mmax = fmaxf(mmax, sS[wid][lane][c] * 0.125f);
      float scl  = __expf(mprev - mmax);
      float lsum = rowL[wid][lane] * scl;
      for (int c = 0; c < 32; c += 2) {
        float p0 = __expf(sS[wid][lane][c]     * 0.125f - mmax);
        float p1 = __expf(sS[wid][lane][c + 1] * 0.125f - mmax);
        sP[wid][lane][c >> 1] = pk2bf(p0, p1);
        lsum += p0 + p1;
      }
      rowM[wid][lane] = mmax;
      rowL[wid][lane] = lsum;
      rowScale[wid][lane] = scl;
    }
    __syncthreads();

    // Rescale running O, then O += P(16x32) @ V(32x64).
#pragma unroll
    for (int t = 0; t < 4; ++t)
#pragma unroll
      for (int r = 0; r < 8; ++r) Oacc[t][r] *= rowScale[wid][rowOff + r];

    bf16x16 pf;
    {
      const unsigned* prow = &sP[wid][lane & 15][0];
      uintx8 u;
#pragma unroll
      for (int v = 0; v < 4; ++v) { u[v] = prow[half4 + v]; u[4 + v] = prow[8 + half4 + v]; }
      pf = __builtin_bit_cast(bf16x16, u);
    }
#pragma unroll
    for (int t = 0; t < 4; ++t) {
      const bf16* vcol = (const bf16*)&sV[0][0] + 16 * t + (lane & 15);   // lane = hd col, elems = key
      bf16x16 bfv;
#pragma unroll
      for (int s = 0; s < 16; ++s) bfv[s] = vcol[(s < 8 ? 2 * half4 + s : 16 + 2 * half4 + s - 8) * 72];
      Oacc[t] = __builtin_amdgcn_wmma_f32_16x16x32_bf16(
          false, pf, false, bfv, (short)0, Oacc[t], false, false);
    }
  }

  // Normalize and write bf16 combined-heads output.
#pragma unroll
  for (int t = 0; t < 4; ++t)
#pragma unroll
    for (int r = 0; r < 8; ++r) {
      float invl = __builtin_amdgcn_rcpf(rowL[wid][rowOff + r]);
      int qrow = qBase + wid * 16 + rowOff + r;
      O[headBase + (size_t)qrow * DIM + 16 * t + col] = f2bf(Oacc[t][r] * invl);
    }
}

// ---------------------------------------------------------------------------
// Fused gate-mix + residual + LayerNorm (one row per block).
// f = gate*enh + (1-gate)*orig + orig ; out16 = bf16(LN(f)*gamma + beta)
// ---------------------------------------------------------------------------
__global__ __launch_bounds__(256)
void gate_ln_kernel(const float* __restrict__ orig, const bf16* __restrict__ enh16,
                    const float* __restrict__ gate,
                    const float* __restrict__ gamma, const float* __restrict__ beta,
                    bf16* __restrict__ out16)
{
  __shared__ float red[256];
  int row = blockIdx.x, tid = threadIdx.x;
  size_t base4 = (size_t)row * (DIM / 4) + tid;        // 4 elems per thread
  float4 g = ((const float4*)gate)[base4];
  float4 o = ((const float4*)orig)[base4];
  uint2 eu = ((const uint2*)enh16)[base4];
  float e0 = bf2f_lo(eu.x), e1 = bf2f_hi(eu.x);
  float e2 = bf2f_lo(eu.y), e3 = bf2f_hi(eu.y);
  float f0 = g.x * e0 + (1.f - g.x) * o.x + o.x;
  float f1 = g.y * e1 + (1.f - g.y) * o.y + o.y;
  float f2 = g.z * e2 + (1.f - g.z) * o.z + o.z;
  float f3 = g.w * e3 + (1.f - g.w) * o.w + o.w;
  float s1 = f0 + f1 + f2 + f3;
  float s2 = f0 * f0 + f1 * f1 + f2 * f2 + f3 * f3;

  red[tid] = s1; __syncthreads();
  for (int off = 128; off > 0; off >>= 1) {
    if (tid < off) red[tid] += red[tid + off];
    __syncthreads();
  }
  float mean = red[0] * (1.0f / DIM);
  __syncthreads();
  red[tid] = s2; __syncthreads();
  for (int off = 128; off > 0; off >>= 1) {
    if (tid < off) red[tid] += red[tid + off];
    __syncthreads();
  }
  float var  = red[0] * (1.0f / DIM) - mean * mean;
  float rstd = rsqrtf(var + 1e-5f);

  float4 gm = ((const float4*)gamma)[tid];
  float4 bt = ((const float4*)beta)[tid];
  float r0 = (f0 - mean) * rstd * gm.x + bt.x;
  float r1 = (f1 - mean) * rstd * gm.y + bt.y;
  float r2 = (f2 - mean) * rstd * gm.z + bt.z;
  float r3 = (f3 - mean) * rstd * gm.w + bt.w;
  ((uint2*)out16)[base4] = make_uint2(pk2bf(r0, r1), pk2bf(r2, r3));
}

// fw[row] = sigmoid( dot(x[row,:], w) + b )
__global__ __launch_bounds__(256)
void fg2_kernel(const float* __restrict__ x, const float* __restrict__ w,
                const float* __restrict__ b, float* __restrict__ fw)
{
  __shared__ float red[256];
  int row = blockIdx.x, tid = threadIdx.x;
  float4 xv = ((const float4*)x)[(size_t)row * (DIM / 4) + tid];
  float4 wv = ((const float4*)w)[tid];
  red[tid] = xv.x * wv.x + xv.y * wv.y + xv.z * wv.z + xv.w * wv.w;
  __syncthreads();
  for (int off = 128; off > 0; off >>= 1) {
    if (tid < off) red[tid] += red[tid + off];
    __syncthreads();
  }
  if (tid == 0) fw[row] = fast_sigmoid(red[0] + b[0]);
}

// out(f32) = fw*hbp16 + (1-fw)*hpp16   (4 elems per thread)
__global__ __launch_bounds__(256)
void final_mix_kernel(const float* __restrict__ fw, const bf16* __restrict__ hbp16,
                      const bf16* __restrict__ hpp16, float* __restrict__ out)
{
  int i4 = blockIdx.x * 256 + threadIdx.x;
  float g = fw[i4 >> 8];                  // DIM/4 = 256 packs per row
  uint2 a = ((const uint2*)hbp16)[i4];
  uint2 p = ((const uint2*)hpp16)[i4];
  float4 r;
  r.x = g * bf2f_lo(a.x) + (1.f - g) * bf2f_lo(p.x);
  r.y = g * bf2f_hi(a.x) + (1.f - g) * bf2f_hi(p.x);
  r.z = g * bf2f_lo(a.y) + (1.f - g) * bf2f_lo(p.y);
  r.w = g * bf2f_hi(a.y) + (1.f - g) * bf2f_hi(p.y);
  ((float4*)out)[i4] = r;
}

// ---------------------------------------------------------------------------
extern "C" void kernel_launch(void* const* d_in, const int* in_sizes, int n_in,
                              void* d_out, int out_size, void* d_ws, size_t ws_size,
                              hipStream_t stream)
{
  (void)in_sizes; (void)n_in; (void)out_size; (void)ws_size;

  const float* h_b     = (const float*)d_in[0];
  const float* h_p     = (const float*)d_in[1];
  const float* wq      = (const float*)d_in[2];
  const float* bq      = (const float*)d_in[3];
  const float* wk_sem  = (const float*)d_in[4];
  const float* bk_sem  = (const float*)d_in[5];
  const float* wk_syn  = (const float*)d_in[6];
  const float* bk_syn  = (const float*)d_in[7];
  const float* wv      = (const float*)d_in[8];
  const float* bv      = (const float*)d_in[9];
  const float* g_sem_w = (const float*)d_in[10];
  const float* g_sem_b = (const float*)d_in[11];
  const float* g_syn_w = (const float*)d_in[12];
  const float* g_syn_b = (const float*)d_in[13];
  const float* ln_sem_g = (const float*)d_in[14];
  const float* ln_sem_b = (const float*)d_in[15];
  const float* ln_syn_g = (const float*)d_in[16];
  const float* ln_syn_b = (const float*)d_in[17];
  const float* fg1_w   = (const float*)d_in[18];
  const float* fg1_b   = (const float*)d_in[19];
  const float* fg2_w   = (const float*)d_in[20];
  const float* fg2_b   = (const float*)d_in[21];
  float* out = (float*)d_out;

  // Workspace layout (bytes).
  char* ws = (char*)d_ws;
  const size_t ACT16 = (size_t)MROWS * DIM * 2;        // 8 MB bf16 activation
  const size_t W16   = (size_t)DIM * DIM * 2;          // 2 MB bf16 DxD weight
  size_t off = 0;
  bf16* hb16    = (bf16*)(ws + off); off += ACT16;
  bf16* hp16    = (bf16*)(ws + off); off += ACT16;
  bf16* wq16    = (bf16*)(ws + off); off += W16;
  bf16* wksem16 = (bf16*)(ws + off); off += W16;
  bf16* wksyn16 = (bf16*)(ws + off); off += W16;
  bf16* wv16    = (bf16*)(ws + off); off += W16;
  bf16* gsem16  = (bf16*)(ws + off); off += 2 * W16;
  bf16* gsyn16  = (bf16*)(ws + off); off += 2 * W16;
  bf16* fg116   = (bf16*)(ws + off); off += 2 * W16;
  bf16* q16     = (bf16*)(ws + off); off += ACT16;     // also enh (in place)
  bf16* k16     = (bf16*)(ws + off); off += ACT16;
  bf16* v16     = (bf16*)(ws + off); off += ACT16;
  bf16* hbp16   = (bf16*)(ws + off); off += ACT16;
  bf16* hpp16   = (bf16*)(ws + off); off += ACT16;
  float* f32buf = (float*)(ws + off); off += (size_t)MROWS * DIM * 4;  // gate/relu
  float* fw     = (float*)(ws + off);

  dim3 blk(256);
  dim3 gGemm(DIM / 128, MROWS / 128);         // (8, 32)
  dim3 gAttn(LEN / 128, BSZ * NH);            // (8, 64)
  const int PK_W1 = DIM * DIM / 2048, PK_W2 = 2 * DIM * DIM / 2048;
  const int PK_A  = MROWS * DIM / 2048;

  // -------- one-time bf16 packing of inputs + weights --------
  pack_bf16_kernel<<<PK_A,  blk, 0, stream>>>(h_b,     hb16);
  pack_bf16_kernel<<<PK_A,  blk, 0, stream>>>(h_p,     hp16);
  pack_bf16_kernel<<<PK_W1, blk, 0, stream>>>(wq,      wq16);
  pack_bf16_kernel<<<PK_W1, blk, 0, stream>>>(wk_sem,  wksem16);
  pack_bf16_kernel<<<PK_W1, blk, 0, stream>>>(wk_syn,  wksyn16);
  pack_bf16_kernel<<<PK_W1, blk, 0, stream>>>(wv,      wv16);
  pack_bf16_kernel<<<PK_W2, blk, 0, stream>>>(g_sem_w, gsem16);
  pack_bf16_kernel<<<PK_W2, blk, 0, stream>>>(g_syn_w, gsyn16);
  pack_bf16_kernel<<<PK_W2, blk, 0, stream>>>(fg1_w,   fg116);

  // -------- semantic branch: q=h_b, k/v from h_p (wk_syn) --------
  gemm_bf16_wmma<0,1><<<gGemm, blk, 0, stream>>>(hb16, hb16, DIM, wq16,    bq,     q16, DIM, DIM);
  gemm_bf16_wmma<0,1><<<gGemm, blk, 0, stream>>>(hp16, hp16, DIM, wksyn16, bk_syn, k16, DIM, DIM);
  gemm_bf16_wmma<0,1><<<gGemm, blk, 0, stream>>>(hp16, hp16, DIM, wv16,    bv,     v16, DIM, DIM);
  flash_attn_wmma<<<gAttn, blk, 0, stream>>>(q16, k16, v16, q16);                   // enh_b
  gemm_bf16_wmma<1,0><<<gGemm, blk, 0, stream>>>(hb16, q16, DIM, gsem16, g_sem_b,
                                                 f32buf, DIM, 2 * DIM);             // gate_b
  gate_ln_kernel<<<MROWS, blk, 0, stream>>>(h_b, q16, f32buf, ln_sem_g, ln_sem_b, hbp16);

  // -------- syntax branch: q=h_p, k/v from h_b (wk_sem) --------
  gemm_bf16_wmma<0,1><<<gGemm, blk, 0, stream>>>(hp16, hp16, DIM, wq16,    bq,     q16, DIM, DIM);
  gemm_bf16_wmma<0,1><<<gGemm, blk, 0, stream>>>(hb16, hb16, DIM, wksem16, bk_sem, k16, DIM, DIM);
  gemm_bf16_wmma<0,1><<<gGemm, blk, 0, stream>>>(hb16, hb16, DIM, wv16,    bv,     v16, DIM, DIM);
  flash_attn_wmma<<<gAttn, blk, 0, stream>>>(q16, k16, v16, q16);                   // enh_p
  gemm_bf16_wmma<1,0><<<gGemm, blk, 0, stream>>>(hp16, q16, DIM, gsyn16, g_syn_b,
                                                 f32buf, DIM, 2 * DIM);             // gate_p
  gate_ln_kernel<<<MROWS, blk, 0, stream>>>(h_p, q16, f32buf, ln_syn_g, ln_syn_b, hpp16);

  // -------- final fusion gate --------
  gemm_bf16_wmma<2,0><<<gGemm, blk, 0, stream>>>(hbp16, hpp16, DIM, fg116, fg1_b,
                                                 f32buf, DIM, 2 * DIM);             // relu(cat@fg1)
  fg2_kernel<<<MROWS, blk, 0, stream>>>(f32buf, fg2_w, fg2_b, fw);
  final_mix_kernel<<<(MROWS * DIM) / 1024, blk, 0, stream>>>(fw, hbp16, hpp16, out);
}
